// MultilayerGRU_20658792694437
// MI455X (gfx1250) — matrix operand order repaired and checked by reading, not compiled, not run on hardware
//
#include <hip/hip_runtime.h>
#include <hip/hip_bf16.h>
#include <math.h>
#include <stdint.h>

#define IN_DIM   512
#define H_DIM    1024
#define OUT_DIM  512
#define NLAYERS  2
#define BB       32
#define SS       1024

typedef __attribute__((ext_vector_type(16))) __bf16 v16bf;
typedef __attribute__((ext_vector_type(8)))  __bf16 v8bf;
typedef __attribute__((ext_vector_type(8)))  float  v8f;
typedef int v4i_gcc __attribute__((vector_size(16)));   // matches builtin param

union V16U { v16bf v; v8bf h[2]; };

#if defined(__gfx1250__) && __has_builtin(__builtin_amdgcn_global_load_async_to_lds_b128)
#define HAS_ASYNC_LDS 1
#endif

// ---------------------------------------------------------------------------
// Async global->LDS 16-byte copy (CDNA5 GLOBAL_LOAD_ASYNC_TO_LDS_B128,
// tracked by ASYNCcnt). Fallback: plain load + LDS store.
__device__ __forceinline__ void copy16_g2l(const __bf16* gsrc, __bf16* lds_dst) {
#if defined(HAS_ASYNC_LDS)
  __builtin_amdgcn_global_load_async_to_lds_b128(
      (__attribute__((address_space(1))) v4i_gcc*)(uintptr_t)gsrc,
      (__attribute__((address_space(3))) v4i_gcc*)(uint32_t)(uintptr_t)lds_dst,
      0, 0);
#else
  *(v8bf*)lds_dst = *(const v8bf*)gsrc;
#endif
}

template <int N>
__device__ __forceinline__ void wait_async() {
#if defined(HAS_ASYNC_LDS)
#if __has_builtin(__builtin_amdgcn_s_wait_asynccnt)
  __builtin_amdgcn_s_wait_asynccnt(N);
#else
  asm volatile("s_wait_asynccnt %0" :: "i"(N) : "memory");
#endif
#endif
}

// ---------------------------------------------------------------------------
// WMMA fragment loaders (layouts per CDNA5 ISA 7.12.2, wave32).
// Work for both global and LDS source pointers (LDS reads lower to ds_load).
// A: 16x32 bf16 (MxK), row-major, stride lda elements.
__device__ __forceinline__ v16bf load_a_frag(const __bf16* __restrict__ A,
                                             int lda, int m0, int k0) {
  int lane = threadIdx.x & 31;
  int r    = lane & 15;
  int kh   = (lane >> 4) << 3;            // 0 or 8
  const __bf16* p = A + (size_t)(m0 + r) * lda + (k0 + kh);
  V16U u;
  u.h[0] = *(const v8bf*)p;               // K = kh .. kh+7
  u.h[1] = *(const v8bf*)(p + 16);        // K = kh+16 .. kh+23
  return u.v;
}

// B: 32x16 (KxN) with B[k][n] = W[n][k]; W row-major NxK, stride ldw.
__device__ __forceinline__ v16bf load_b_frag(const __bf16* __restrict__ W,
                                             int ldw, int n0, int k0) {
  int lane = threadIdx.x & 31;
  int n    = lane & 15;
  int kh   = (lane >> 4) << 3;            // 0 or 8
  const __bf16* p = W + (size_t)(n0 + n) * ldw + (k0 + kh);
  V16U u;
  u.h[0] = *(const v8bf*)p;
  u.h[1] = *(const v8bf*)(p + 16);
  return u.v;
}

__device__ __forceinline__ v8f wmma_bf16(v16bf a, v16bf b, v8f c) {
  return __builtin_amdgcn_wmma_f32_16x16x32_bf16(false, a, false, b,
                                                 (short)0, c, false, false);
}

__device__ __forceinline__ v8f zero8() {
  v8f z = {0.f, 0.f, 0.f, 0.f, 0.f, 0.f, 0.f, 0.f};
  return z;
}

// ---------------------------------------------------------------------------
__global__ void __launch_bounds__(256) cvt_f32_bf16_kernel(
    const float* __restrict__ s, __bf16* __restrict__ d, long n) {
  long i      = (long)blockIdx.x * blockDim.x + threadIdx.x;
  long stride = (long)gridDim.x * blockDim.x;
  for (; i < n; i += stride) d[i] = (__bf16)s[i];
}

// ---------------------------------------------------------------------------
// C[M x N] = A[M x K](bf16) * W[N x K]^T(bf16) + bias[N], fp32 accumulate.
// Block: 256 thr = 8 waves (2 M x 4 N), wave tile 32x32 (2x2 frags).
// Block tile 64 x 128, K-step 32, LDS double-buffered with async copies.
// Requires M%64==0, N%128==0, K%32==0.
#define GBM 64
#define GBN 128
#define GBK 32
#define LPAD 40   // LDS row stride (elements): 80B, spreads banks
__global__ void __launch_bounds__(256) gemm_bf16_bias_kernel(
    const __bf16* __restrict__ A, const __bf16* __restrict__ W,
    const float* __restrict__ bias, float* __restrict__ C,
    int M, int N, int K) {
  __shared__ __bf16 sA[2][GBM * LPAD];    // 2 x 5120 B
  __shared__ __bf16 sB[2][GBN * LPAD];    // 2 x 10240 B

  int tid  = threadIdx.x;
  int wave = tid >> 5, lane = tid & 31;
  int wm = wave >> 2, wn = wave & 3;
  int mBase = blockIdx.y * GBM;
  int nBase = blockIdx.x * GBN;

  // Cooperative tile copy: 16B chunks. A: 64 rows x 4 chunks = 256 (1/thr).
  // B: 128 rows x 4 chunks = 512 (2/thr).
  int crow = tid >> 2;
  int ccol = (tid & 3) * 8;

  auto issue = [&](int buf, int k0) {
    copy16_g2l(A + (size_t)(mBase + crow) * K + k0 + ccol,
               &sA[buf][crow * LPAD + ccol]);
    copy16_g2l(W + (size_t)(nBase + crow) * K + k0 + ccol,
               &sB[buf][crow * LPAD + ccol]);
    copy16_g2l(W + (size_t)(nBase + crow + 64) * K + k0 + ccol,
               &sB[buf][(crow + 64) * LPAD + ccol]);
  };

  v8f acc[2][2] = {{zero8(), zero8()}, {zero8(), zero8()}};

  issue(0, 0);
  int KT = K / GBK;
  for (int kt = 0; kt < KT; ++kt) {
    int cur = kt & 1;
    if (kt + 1 < KT) {
      issue(cur ^ 1, (kt + 1) * GBK);     // overlap next copy with compute
      wait_async<3>();                    // prev 3 chunks (cur) have landed
    } else {
      wait_async<0>();
    }
    __syncthreads();                      // cur tile visible to all waves

    v16bf afrag[2], bfrag[2];
#pragma unroll
    for (int i = 0; i < 2; ++i)
      afrag[i] = load_a_frag(sA[cur], LPAD, wm * 32 + i * 16, 0);
#pragma unroll
    for (int j = 0; j < 2; ++j)
      bfrag[j] = load_b_frag(sB[cur], LPAD, wn * 32 + j * 16, 0);
#pragma unroll
    for (int i = 0; i < 2; ++i)
#pragma unroll
      for (int j = 0; j < 2; ++j)
        acc[i][j] = wmma_bf16(afrag[i], bfrag[j], acc[i][j]);

    __syncthreads();                      // done reading cur before reuse
  }

  int row_hi = (lane >> 4) << 3;          // 0 or 8
  int colL   = lane & 15;
#pragma unroll
  for (int i = 0; i < 2; ++i) {
#pragma unroll
    for (int j = 0; j < 2; ++j) {
      int n    = nBase + wn * 32 + j * 16 + colL;
      float bv = bias[n];
#pragma unroll
      for (int r = 0; r < 8; ++r) {
        int m = mBase + wm * 32 + i * 16 + r + row_hi;
        C[(size_t)m * N + n] = acc[i][j][r] + bv;
      }
    }
  }
}

// ---------------------------------------------------------------------------
// GRU step, phase 1: z, r gates (weights stream from L2; M=32 tiny).
__global__ void __launch_bounds__(256) gru_zr_step_kernel(
    const __bf16* __restrict__ hbf, const float* __restrict__ hf,
    const __bf16* __restrict__ Wh,      // (3H, H) bf16, gate 0 = z, 1 = r
    const float* __restrict__ pre,      // (B, S, 3, H) fp32
    int t, float* __restrict__ zbuf, __bf16* __restrict__ rhbuf) {
  int wave = threadIdx.x >> 5;
  int lane = threadIdx.x & 31;
  int tid  = blockIdx.x * 8 + wave;     // 0..255
  int gate = tid >> 7;                  // 0 = z, 1 = r
  int rem  = tid & 127;
  int m0   = (rem >> 6) << 4;
  int n0   = (rem & 63) << 4;
  const __bf16* Wg = Wh + (size_t)gate * H_DIM * H_DIM;

  v8f acc = zero8();
  for (int k0 = 0; k0 < H_DIM; k0 += 32) {
    v16bf a = load_a_frag(hbf, H_DIM, m0, k0);
    v16bf b = load_b_frag(Wg, H_DIM, n0, k0);
    acc = wmma_bf16(a, b, acc);
  }

  int row_hi = (lane >> 4) << 3;
  int col    = n0 + (lane & 15);
#pragma unroll
  for (int j = 0; j < 8; ++j) {
    int b = m0 + j + row_hi;
    size_t pidx = (((size_t)b * SS + t) * 3 + gate) * H_DIM + col;
    float v = acc[j] + pre[pidx];
    float s = 1.0f / (1.0f + __expf(-v));
    if (gate == 0) {
      zbuf[b * H_DIM + col] = s;
    } else {
      rhbuf[b * H_DIM + col] = (__bf16)(s * hf[b * H_DIM + col]);
    }
  }
}

// GRU step, phase 2: g gate + state update.
__global__ void __launch_bounds__(256) gru_g_step_kernel(
    const __bf16* __restrict__ rhbf,
    const __bf16* __restrict__ Wg,      // gate-2 slice (H, H) bf16
    const float* __restrict__ pre, int t,
    const float* __restrict__ zbuf,
    float* __restrict__ hf, __bf16* __restrict__ hbf,
    __bf16* __restrict__ y) {
  int wave = threadIdx.x >> 5;
  int lane = threadIdx.x & 31;
  int tid  = blockIdx.x * 8 + wave;     // 0..127
  int m0   = (tid >> 6) << 4;
  int n0   = (tid & 63) << 4;

  v8f acc = zero8();
  for (int k0 = 0; k0 < H_DIM; k0 += 32) {
    v16bf a = load_a_frag(rhbf, H_DIM, m0, k0);
    v16bf b = load_b_frag(Wg, H_DIM, n0, k0);
    acc = wmma_bf16(a, b, acc);
  }

  int row_hi = (lane >> 4) << 3;
  int col    = n0 + (lane & 15);
#pragma unroll
  for (int j = 0; j < 8; ++j) {
    int b = m0 + j + row_hi;
    size_t pidx = (((size_t)b * SS + t) * 3 + 2) * H_DIM + col;
    float g  = tanhf(acc[j] + pre[pidx]);
    float z  = zbuf[b * H_DIM + col];
    float hn = z * hf[b * H_DIM + col] + (1.0f - z) * g;
    hf[b * H_DIM + col]  = hn;
    hbf[b * H_DIM + col] = (__bf16)hn;
    y[((size_t)b * SS + t) * H_DIM + col] = (__bf16)hn;
  }
}

// ---------------------------------------------------------------------------
__global__ void __launch_bounds__(256) init_h_kernel(
    const float* __restrict__ h0, int layer,
    float* __restrict__ hf, __bf16* __restrict__ hbf) {
  int i = blockIdx.x * blockDim.x + threadIdx.x;    // 0 .. 32767
  int b = i >> 10, n = i & (H_DIM - 1);
  float v = h0[((size_t)b * NLAYERS + layer) * H_DIM + n];
  hf[i]  = v;
  hbf[i] = (__bf16)v;
}

__global__ void __launch_bounds__(256) write_hidden_kernel(
    const float* __restrict__ hf0, const float* __restrict__ hf1,
    float* __restrict__ out) {
  int i = blockIdx.x * blockDim.x + threadIdx.x;    // 0 .. 65535
  int b = i >> 11;
  int l = (i >> 10) & 1;
  int n = i & (H_DIM - 1);
  out[i] = (l == 0 ? hf0 : hf1)[b * H_DIM + n];
}

// ---------------------------------------------------------------------------
extern "C" void kernel_launch(void* const* d_in, const int* in_sizes, int n_in,
                              void* d_out, int out_size, void* d_ws, size_t ws_size,
                              hipStream_t stream) {
  (void)in_sizes; (void)n_in; (void)out_size; (void)ws_size;
  const float* x   = (const float*)d_in[0];
  const float* h0  = (const float*)d_in[1];
  const float* Wx0 = (const float*)d_in[2];
  const float* bx0 = (const float*)d_in[3];
  const float* Wh0 = (const float*)d_in[4];
  const float* Wx1 = (const float*)d_in[5];
  const float* bx1 = (const float*)d_in[6];
  const float* Wh1 = (const float*)d_in[7];
  const float* Why = (const float*)d_in[8];
  const float* bhy = (const float*)d_in[9];
  float* out = (float*)d_out;

  const int MX = BB * SS;   // 32768 rows for the batched GEMMs

  size_t off = 0;
  auto alloc = [&](size_t bytes) -> void* {
    void* p = (char*)d_ws + off;
    off += (bytes + 255) & ~(size_t)255;
    return p;
  };
  __bf16* Xbf   = (__bf16*)alloc((size_t)MX * IN_DIM * 2);
  __bf16* Wx0b  = (__bf16*)alloc((size_t)3 * H_DIM * IN_DIM * 2);
  __bf16* Wh0b  = (__bf16*)alloc((size_t)3 * H_DIM * H_DIM * 2);
  __bf16* Wx1b  = (__bf16*)alloc((size_t)3 * H_DIM * H_DIM * 2);
  __bf16* Wh1b  = (__bf16*)alloc((size_t)3 * H_DIM * H_DIM * 2);
  __bf16* Whyb  = (__bf16*)alloc((size_t)OUT_DIM * H_DIM * 2);
  float*  pre   = (float*)alloc((size_t)MX * 3 * H_DIM * 4);   // reused per layer
  __bf16* y0    = (__bf16*)alloc((size_t)MX * H_DIM * 2);
  __bf16* y1    = (__bf16*)alloc((size_t)MX * H_DIM * 2);
  float*  hf0   = (float*)alloc((size_t)BB * H_DIM * 4);
  float*  hf1   = (float*)alloc((size_t)BB * H_DIM * 4);
  __bf16* hb0   = (__bf16*)alloc((size_t)BB * H_DIM * 2);
  __bf16* hb1   = (__bf16*)alloc((size_t)BB * H_DIM * 2);
  float*  zbuf  = (float*)alloc((size_t)BB * H_DIM * 4);
  __bf16* rhbuf = (__bf16*)alloc((size_t)BB * H_DIM * 2);

  // --- fp32 -> bf16 conversions -------------------------------------------
  cvt_f32_bf16_kernel<<<4096, 256, 0, stream>>>(x,   Xbf,  (long)MX * IN_DIM);
  cvt_f32_bf16_kernel<<<1024, 256, 0, stream>>>(Wx0, Wx0b, (long)3 * H_DIM * IN_DIM);
  cvt_f32_bf16_kernel<<<2048, 256, 0, stream>>>(Wh0, Wh0b, (long)3 * H_DIM * H_DIM);
  cvt_f32_bf16_kernel<<<2048, 256, 0, stream>>>(Wx1, Wx1b, (long)3 * H_DIM * H_DIM);
  cvt_f32_bf16_kernel<<<2048, 256, 0, stream>>>(Wh1, Wh1b, (long)3 * H_DIM * H_DIM);
  cvt_f32_bf16_kernel<<<512,  256, 0, stream>>>(Why, Whyb, (long)OUT_DIM * H_DIM);

  // --- layer 0 -------------------------------------------------------------
  gemm_bf16_bias_kernel<<<dim3((3 * H_DIM) / GBN, MX / GBM), 256, 0, stream>>>(
      Xbf, Wx0b, bx0, pre, MX, 3 * H_DIM, IN_DIM);
  init_h_kernel<<<(BB * H_DIM) / 256, 256, 0, stream>>>(h0, 0, hf0, hb0);
  for (int t = 0; t < SS; ++t) {
    gru_zr_step_kernel<<<32, 256, 0, stream>>>(hb0, hf0, Wh0b, pre, t, zbuf, rhbuf);
    gru_g_step_kernel<<<16, 256, 0, stream>>>(rhbuf,
        Wh0b + (size_t)2 * H_DIM * H_DIM, pre, t, zbuf, hf0, hb0, y0);
  }

  // --- layer 1 -------------------------------------------------------------
  gemm_bf16_bias_kernel<<<dim3((3 * H_DIM) / GBN, MX / GBM), 256, 0, stream>>>(
      y0, Wx1b, bx1, pre, MX, 3 * H_DIM, H_DIM);
  init_h_kernel<<<(BB * H_DIM) / 256, 256, 0, stream>>>(h0, 1, hf1, hb1);
  for (int t = 0; t < SS; ++t) {
    gru_zr_step_kernel<<<32, 256, 0, stream>>>(hb1, hf1, Wh1b, pre, t, zbuf, rhbuf);
    gru_g_step_kernel<<<16, 256, 0, stream>>>(rhbuf,
        Wh1b + (size_t)2 * H_DIM * H_DIM, pre, t, zbuf, hf1, hb1, y1);
  }

  // --- output projection + hidden ------------------------------------------
  gemm_bf16_bias_kernel<<<dim3(OUT_DIM / GBN, MX / GBM), 256, 0, stream>>>(
      y1, Whyb, bhy, out, MX, OUT_DIM, H_DIM);
  write_hidden_kernel<<<(BB * NLAYERS * H_DIM) / 256, 256, 0, stream>>>(
      hf0, hf1, out + (size_t)MX * OUT_DIM);
}
